// ViT_4260607557883
// MI455X (gfx1250) — matrix operand order repaired and checked by reading, not compiled, not run on hardware
//
#include <hip/hip_runtime.h>
#include <hip/hip_bf16.h>
#include <math.h>

// ---------------- constants (match reference) ----------------
constexpr int H_   = 240;
constexpr int W_   = 240;
constexpr int PH_  = 16;
constexpr int PW_  = 16;
constexpr int VP_  = 15;
constexpr int HP_  = 15;
constexpr int NP_  = 225;
constexpr int DIM_ = 512;
constexpr int MLP_ = 2048;
constexpr int PD_  = 768;   // 16*16*3
constexpr int B_   = 256;
constexpr int T_   = 20;
constexpr int NTOK = B_ * NP_;   // 57600
constexpr int KV_  = 2 * DIM_;   // fused K|V projection width

typedef __attribute__((ext_vector_type(16))) __bf16 v16bf;
typedef __attribute__((ext_vector_type(8)))  __bf16 v8bf;
typedef __attribute__((ext_vector_type(8)))  float  v8f;

// ---------------- generic bf16 WMMA GEMM (double-buffered) ----------------
// C[M,N] = A[M,K] @ Bt[N,K]^T  (Bt pre-transposed: row = output column)
// MODE 0: Cf = acc
// MODE 1: Cf = acc + bias
// MODE 2: Cb = bf16(acc)
// MODE 3: Cb = bf16(gelu(acc + bias))          (exact erf gelu)
// MODE 4: x = Cf + acc + bias; Cf = x; Cb = bf16(x)   (residual)
// Requires: M % 128 == 0, N % 128 == 0, K % 64 == 0.
template<int MODE>
__global__ __launch_bounds__(256) void gemm_bf16_wmma(
    const __hip_bfloat16* __restrict__ A,
    const __hip_bfloat16* __restrict__ Bt,
    const float* __restrict__ bias,
    float* __restrict__ Cf,
    __hip_bfloat16* __restrict__ Cb,
    int M, int N, int K)
{
  const int lane = threadIdx.x & 31;
  const int wave = threadIdx.x >> 5;
  const int wm = wave & 1;            // 2 waves along M
  const int wn = wave >> 1;           // 4 waves along N
  const int m0 = blockIdx.y * 128 + wm * 64;   // 4 x 16 rows
  const int n0 = blockIdx.x * 128 + wn * 32;   // 2 x 16 cols
  const int lm = lane & 15;
  const int lh = lane >> 4;

  v8f acc[4][2];
#pragma unroll
  for (int i = 0; i < 4; ++i)
#pragma unroll
    for (int j = 0; j < 2; ++j)
#pragma unroll
      for (int e = 0; e < 8; ++e) acc[i][j][e] = 0.0f;

  const __bf16* Ab = (const __bf16*)A;
  const __bf16* Bb = (const __bf16*)Bt;

  // 16-bit A layout: lanes 0-15 hold K [0..7] and [16..23]; lanes 16-31 hold [8..15] and [24..31]
  auto loadA = [&](int kb, v16bf (&af)[4]) {
#pragma unroll
    for (int i = 0; i < 4; ++i) {
      const __bf16* p = Ab + (size_t)(m0 + i * 16 + lm) * K + kb + lh * 8;
      v8bf lo = *(const v8bf*)p;
      v8bf hi = *(const v8bf*)(p + 16);
#pragma unroll
      for (int e = 0; e < 8; ++e) { af[i][e] = lo[e]; af[i][e + 8] = hi[e]; }
    }
  };
  // B layout: lane (0-15) = column, lanes 0-15 hold K [0..7] and [16..23]; lanes 16-31 hold [8..15] and [24..31]
  auto loadB = [&](int kb, v16bf (&bf)[2]) {
#pragma unroll
    for (int j = 0; j < 2; ++j) {
      const __bf16* p = Bb + (size_t)(n0 + j * 16 + lm) * K + kb + lh * 8;
      v8bf lo = *(const v8bf*)p;
      v8bf hi = *(const v8bf*)(p + 16);
#pragma unroll
      for (int e = 0; e < 8; ++e) { bf[j][e] = lo[e]; bf[j][e + 8] = hi[e]; }
    }
  };
  auto mma = [&](v16bf (&af)[4], v16bf (&bf)[2]) {
#pragma unroll
    for (int i = 0; i < 4; ++i)
#pragma unroll
      for (int j = 0; j < 2; ++j)
        acc[i][j] = __builtin_amdgcn_wmma_f32_16x16x32_bf16(
            false, af[i], false, bf[j], (short)0, acc[i][j], false, false);
  };

  v16bf A0[4], A1[4], B0[2], B1[2];
  loadA(0, A0);  loadB(0, B0);
  loadA(32, A1); loadB(32, B1);
  for (int kb = 64; kb < K; kb += 64) {
    // cache prefetch two K-steps ahead (speculative; emits global_prefetch_b8)
    {
      const int kp = kb + 64;
#pragma unroll
      for (int i = 0; i < 4; ++i)
        __builtin_prefetch(Ab + (size_t)(m0 + i * 16 + lm) * K + kp + lh * 8, 0, 1);
#pragma unroll
      for (int j = 0; j < 2; ++j)
        __builtin_prefetch(Bb + (size_t)(n0 + j * 16 + lm) * K + kp + lh * 8, 0, 1);
    }
    mma(A0, B0);
    loadA(kb, A0);       loadB(kb, B0);
    mma(A1, B1);
    loadA(kb + 32, A1);  loadB(kb + 32, B1);
  }
  mma(A0, B0);
  mma(A1, B1);

#pragma unroll
  for (int i = 0; i < 4; ++i) {
#pragma unroll
    for (int j = 0; j < 2; ++j) {
      const int col = n0 + j * 16 + lm;
      float bv = 0.0f;
      if constexpr (MODE == 1 || MODE == 3 || MODE == 4) bv = bias[col];
#pragma unroll
      for (int r = 0; r < 8; ++r) {
        const int row = m0 + i * 16 + r + lh * 8;   // C/D: vgpr r -> M=r (lanes 0-15), M=r+8 (lanes 16-31)
        const size_t o = (size_t)row * N + col;
        float v = acc[i][j][r] + bv;
        if constexpr (MODE == 0 || MODE == 1) {
          Cf[o] = v;
        } else if constexpr (MODE == 2) {
          Cb[o] = __float2bfloat16(v);
        } else if constexpr (MODE == 3) {
          float g = 0.5f * v * (1.0f + erff(v * 0.70710678118654752f));
          Cb[o] = __float2bfloat16(g);
        } else {  // MODE == 4
          float x = Cf[o] + v;
          Cf[o] = x;
          Cb[o] = __float2bfloat16(x);
        }
      }
    }
  }
}

// ---------------- weight transpose f32 (K,N) -> bf16 (N,K) ----------------
// dst layer z starts at Wt + z*zStrideDst (elements); src layer z at W + z*K*N.
__global__ void transpose_to_bf16(const float* __restrict__ W,
                                  __hip_bfloat16* __restrict__ Wt,
                                  int K, int N, int zStrideDst)
{
  const int mat = K * N;
  const float* src = W + (size_t)blockIdx.z * mat;
  __hip_bfloat16* dst = Wt + (size_t)blockIdx.z * zStrideDst;
  for (int i = blockIdx.x * 256 + threadIdx.x; i < mat; i += gridDim.x * 256) {
    const int k = i / N;
    const int n = i - k * N;
    dst[(size_t)n * K + k] = __float2bfloat16(src[i]);
  }
}

// ---------------- patch gather + LN1 -> bf16 (NTOK x 768) ----------------
__global__ __launch_bounds__(256) void patch_embed_ln(
    const float* __restrict__ bev, const int* __restrict__ av, int S,
    const float* __restrict__ g, const float* __restrict__ bt,
    __hip_bfloat16* __restrict__ out)
{
  __shared__ float red[256];
  const int t = blockIdx.x;
  const int b = t / NP_, p = t % NP_;
  int sid = 0;
  for (int i = 0; i < S; ++i) if (av[i] <= b) sid = i;
  const int vp = p / HP_, hp = p % HP_;
  const int tid = threadIdx.x;

  float loc[3];
  float s = 0.0f;
#pragma unroll
  for (int q = 0; q < 3; ++q) {
    const int d = tid + q * 256;
    const int ph = d / 48, rem = d % 48, pw = rem / 3, c = rem % 3;
    const int row = vp * PH_ + ph, col = hp * PW_ + pw;
    const float v = bev[(((size_t)sid * H_ + row) * W_ + col) * 3 + c];
    loc[q] = v; s += v;
  }
  red[tid] = s; __syncthreads();
  for (int o = 128; o > 0; o >>= 1) { if (tid < o) red[tid] += red[tid + o]; __syncthreads(); }
  const float mu = red[0] * (1.0f / 768.0f); __syncthreads();
  float s2 = 0.0f;
#pragma unroll
  for (int q = 0; q < 3; ++q) { const float d0 = loc[q] - mu; s2 += d0 * d0; }
  red[tid] = s2; __syncthreads();
  for (int o = 128; o > 0; o >>= 1) { if (tid < o) red[tid] += red[tid + o]; __syncthreads(); }
  const float rstd = rsqrtf(red[0] * (1.0f / 768.0f) + 1e-5f);
#pragma unroll
  for (int q = 0; q < 3; ++q) {
    const int d = tid + q * 256;
    out[(size_t)t * PD_ + d] = __float2bfloat16((loc[q] - mu) * rstd * g[d] + bt[d]);
  }
}

// ---------------- LN2 (in place on x) + pos_emb, write f32 + bf16 ----------------
__global__ __launch_bounds__(256) void ln2_pos(
    float* __restrict__ x, __hip_bfloat16* __restrict__ xb,
    const float* __restrict__ g, const float* __restrict__ bt,
    const float* __restrict__ pos)
{
  __shared__ float red[256];
  const int t = blockIdx.x;
  const int p = t % NP_;
  const int tid = threadIdx.x;
  float* row = x + (size_t)t * DIM_;
  const float v0 = row[tid], v1 = row[tid + 256];
  red[tid] = v0 + v1; __syncthreads();
  for (int o = 128; o > 0; o >>= 1) { if (tid < o) red[tid] += red[tid + o]; __syncthreads(); }
  const float mu = red[0] * (1.0f / 512.0f); __syncthreads();
  const float d0 = v0 - mu, d1 = v1 - mu;
  red[tid] = d0 * d0 + d1 * d1; __syncthreads();
  for (int o = 128; o > 0; o >>= 1) { if (tid < o) red[tid] += red[tid + o]; __syncthreads(); }
  const float rstd = rsqrtf(red[0] * (1.0f / 512.0f) + 1e-5f);
  const float o0 = d0 * rstd * g[tid] + bt[tid] + pos[(size_t)p * DIM_ + tid];
  const float o1 = d1 * rstd * g[tid + 256] + bt[tid + 256] + pos[(size_t)p * DIM_ + tid + 256];
  row[tid] = o0; row[tid + 256] = o1;
  xb[(size_t)t * DIM_ + tid] = __float2bfloat16(o0);
  xb[(size_t)t * DIM_ + tid + 256] = __float2bfloat16(o1);
}

// ---------------- per-layer FFN pre-LN: x f32 -> h bf16 ----------------
__global__ __launch_bounds__(256) void ln_to_bf16(
    const float* __restrict__ x, __hip_bfloat16* __restrict__ hb,
    const float* __restrict__ g, const float* __restrict__ bt)
{
  __shared__ float red[256];
  const int t = blockIdx.x;
  const int tid = threadIdx.x;
  const float* row = x + (size_t)t * DIM_;
  const float v0 = row[tid], v1 = row[tid + 256];
  red[tid] = v0 + v1; __syncthreads();
  for (int o = 128; o > 0; o >>= 1) { if (tid < o) red[tid] += red[tid + o]; __syncthreads(); }
  const float mu = red[0] * (1.0f / 512.0f); __syncthreads();
  const float d0 = v0 - mu, d1 = v1 - mu;
  red[tid] = d0 * d0 + d1 * d1; __syncthreads();
  for (int o = 128; o > 0; o >>= 1) { if (tid < o) red[tid] += red[tid + o]; __syncthreads(); }
  const float rstd = rsqrtf(red[0] * (1.0f / 512.0f) + 1e-5f);
  hb[(size_t)t * DIM_ + tid]       = __float2bfloat16(d0 * rstd * g[tid] + bt[tid]);
  hb[(size_t)t * DIM_ + tid + 256] = __float2bfloat16(d1 * rstd * g[tid + 256] + bt[tid + 256]);
}

// ---------------- patch index from positions (matches _patch_idx) ----------------
__device__ __forceinline__ int patch_index(const float* __restrict__ positions, int b)
{
  const float p0 = positions[(size_t)b * T_ * 2 + 19 * 2 + 0];
  const float p1 = positions[(size_t)b * T_ * 2 + 19 * 2 + 1];
  const float rx = -p1, ry = p0;
  const float sx = rx * 8.0f + 120.0f;    // W/30, W/2
  const float sy = 120.0f - ry * 4.0f;    // H/2, H/60
  float r = floorf(sy * (1.0f / 16.0f));  r = fminf(fmaxf(r, 0.0f), 14.0f);
  float c = floorf(sx * (1.0f / 16.0f));  c = fminf(fmaxf(c, 0.0f), 14.0f);
  return (int)r * HP_ + (int)c;
}

// gather selected token row (f32 x -> bf16), one block per batch element
__global__ __launch_bounds__(256) void gather_sel(
    const float* __restrict__ x, const float* __restrict__ positions,
    __hip_bfloat16* __restrict__ sel)
{
  const int b = blockIdx.x;
  const int idx = patch_index(positions, b);
  const float* row = x + ((size_t)b * NP_ + idx) * DIM_;
  const int tid = threadIdx.x;
  sel[(size_t)b * DIM_ + tid]       = __float2bfloat16(row[tid]);
  sel[(size_t)b * DIM_ + tid + 256] = __float2bfloat16(row[tid + 256]);
}

// ---------------- attention: one block per (b, head); K/V interleaved stride 1024 ----------------
__global__ __launch_bounds__(256) void attn_kernel(
    const float* __restrict__ q,             // B x 512
    const __hip_bfloat16* __restrict__ kv,   // NTOK x 1024 : [K | V]
    float* __restrict__ out)                 // B x 512
{
  __shared__ float qs[64];
  __shared__ float dots[256];
  __shared__ float red[256];
  const int b = blockIdx.x, h = blockIdx.y;
  const int tid = threadIdx.x;
  if (tid < 64) qs[tid] = q[(size_t)b * DIM_ + h * 64 + tid];
  __syncthreads();

  float d = -INFINITY;
  if (tid < NP_) {
    const __hip_bfloat16* kp = kv + ((size_t)b * NP_ + tid) * KV_ + h * 64;
    float s = 0.0f;
#pragma unroll 8
    for (int e = 0; e < 64; ++e) s += qs[e] * __bfloat162float(kp[e]);
    d = s * 0.125f;   // DH^-0.5
  }
  red[tid] = d; __syncthreads();
  for (int o = 128; o > 0; o >>= 1) { if (tid < o) red[tid] = fmaxf(red[tid], red[tid + o]); __syncthreads(); }
  const float mx = red[0]; __syncthreads();
  const float e = (tid < NP_) ? __expf(d - mx) : 0.0f;
  dots[tid] = e; red[tid] = e; __syncthreads();
  for (int o = 128; o > 0; o >>= 1) { if (tid < o) red[tid] += red[tid + o]; __syncthreads(); }
  const float inv = 1.0f / red[0]; __syncthreads();

  const int dd = tid & 63, grp = tid >> 6;   // 4 groups of 64 lanes split NP
  float acc = 0.0f;
  for (int n = grp; n < NP_; n += 4)
    acc += dots[n] * __bfloat162float(kv[((size_t)b * NP_ + n) * KV_ + DIM_ + h * 64 + dd]);
  red[tid] = acc * inv; __syncthreads();
  if (tid < 64)
    out[(size_t)b * DIM_ + h * 64 + tid] =
        red[tid] + red[tid + 64] + red[tid + 128] + red[tid + 192];
}

__global__ void f32_to_bf16_kernel(const float* __restrict__ in,
                                   __hip_bfloat16* __restrict__ out, int n)
{
  const int i = blockIdx.x * 256 + threadIdx.x;
  if (i < n) out[i] = __float2bfloat16(in[i]);
}

__global__ __launch_bounds__(256) void scatter_add(
    float* __restrict__ x, const float* __restrict__ proj,
    const float* __restrict__ positions)
{
  const int b = blockIdx.x;
  const int idx = patch_index(positions, b);
  float* row = x + ((size_t)b * NP_ + idx) * DIM_;
  const float* pr = proj + (size_t)b * DIM_;
  row[threadIdx.x] += pr[threadIdx.x];
  row[threadIdx.x + 256] += pr[threadIdx.x + 256];
}

// ---------------- final LN(x[:,0]) + head ----------------
__global__ __launch_bounds__(128) void head_kernel(
    const float* __restrict__ x, const float* __restrict__ g,
    const float* __restrict__ bt, const float* __restrict__ hw,
    const float* __restrict__ hb, float* __restrict__ out)
{
  __shared__ float sb[512];
  __shared__ float red[128];
  const int b = blockIdx.x, tid = threadIdx.x;
  const float* row = x + (size_t)b * NP_ * DIM_;   // token 0
  float l[4]; float s = 0.0f;
#pragma unroll
  for (int q = 0; q < 4; ++q) { l[q] = row[tid + q * 128]; s += l[q]; }
  red[tid] = s; __syncthreads();
  for (int o = 64; o > 0; o >>= 1) { if (tid < o) red[tid] += red[tid + o]; __syncthreads(); }
  const float mu = red[0] * (1.0f / 512.0f); __syncthreads();
  float s2 = 0.0f;
#pragma unroll
  for (int q = 0; q < 4; ++q) { const float d = l[q] - mu; s2 += d * d; }
  red[tid] = s2; __syncthreads();
  for (int o = 64; o > 0; o >>= 1) { if (tid < o) red[tid] += red[tid + o]; __syncthreads(); }
  const float rstd = rsqrtf(red[0] * (1.0f / 512.0f) + 1e-5f);
#pragma unroll
  for (int q = 0; q < 4; ++q) {
    const int d = tid + q * 128;
    sb[d] = (l[q] - mu) * rstd * g[d] + bt[d];
  }
  __syncthreads();
  if (tid < 10) {
    float acc = hb[tid];
    for (int e = 0; e < 512; ++e) acc += sb[e] * hw[e * 10 + tid];
    out[(size_t)b * 10 + tid] = acc;
  }
}

// ---------------- launch ----------------
extern "C" void kernel_launch(void* const* d_in, const int* in_sizes, int n_in,
                              void* d_out, int out_size, void* d_ws, size_t ws_size,
                              hipStream_t stream) {
  const float* bev      = (const float*)d_in[0];
  const int*   av       = (const int*)d_in[1];
  const int    S        = in_sizes[1];
  const float* positions= (const float*)d_in[2];
  const float* pe_ln1_g = (const float*)d_in[3];
  const float* pe_ln1_b = (const float*)d_in[4];
  const float* pe_w     = (const float*)d_in[5];
  const float* pe_b     = (const float*)d_in[6];
  const float* pe_ln2_g = (const float*)d_in[7];
  const float* pe_ln2_b = (const float*)d_in[8];
  const float* pos_emb  = (const float*)d_in[9];
  const float* wq       = (const float*)d_in[10];
  const float* wk       = (const float*)d_in[11];
  const float* wv       = (const float*)d_in[12];
  const float* wo       = (const float*)d_in[13];
  const float* bo       = (const float*)d_in[14];
  const float* ff_ln_g  = (const float*)d_in[15];
  const float* ff_ln_b  = (const float*)d_in[16];
  const float* w1       = (const float*)d_in[17];
  const float* b1       = (const float*)d_in[18];
  const float* w2       = (const float*)d_in[19];
  const float* b2       = (const float*)d_in[20];
  const float* norm_g   = (const float*)d_in[21];
  const float* norm_b   = (const float*)d_in[22];
  const float* head_w   = (const float*)d_in[23];
  const float* head_b   = (const float*)d_in[24];
  float* out = (float*)d_out;

  // workspace layout
  char* ws = (char*)d_ws;
  size_t off = 0;
  auto alloc = [&](size_t bytes) { size_t o = off; off = (off + bytes + 255) & ~(size_t)255; return o; };
  float*          X    = (float*)(ws + alloc((size_t)NTOK * DIM_ * 4));
  __hip_bfloat16* XB   = (__hip_bfloat16*)(ws + alloc((size_t)NTOK * DIM_ * 2));
  __hip_bfloat16* KVB  = (__hip_bfloat16*)(ws + alloc((size_t)NTOK * KV_ * 2));  // K|V, later FFN h
  __hip_bfloat16* GB   = (__hip_bfloat16*)(ws + alloc((size_t)NTOK * MLP_ * 2)); // also patches
  __hip_bfloat16* peT  = (__hip_bfloat16*)(ws + alloc((size_t)PD_ * DIM_ * 2));
  __hip_bfloat16* wqT  = (__hip_bfloat16*)(ws + alloc((size_t)4 * DIM_ * DIM_ * 2));
  __hip_bfloat16* wkvT = (__hip_bfloat16*)(ws + alloc((size_t)4 * KV_ * DIM_ * 2));
  __hip_bfloat16* woT  = (__hip_bfloat16*)(ws + alloc((size_t)4 * DIM_ * DIM_ * 2));
  __hip_bfloat16* w1T  = (__hip_bfloat16*)(ws + alloc((size_t)4 * DIM_ * MLP_ * 2));
  __hip_bfloat16* w2T  = (__hip_bfloat16*)(ws + alloc((size_t)4 * MLP_ * DIM_ * 2));
  float*          QF   = (float*)(ws + alloc((size_t)B_ * DIM_ * 4));
  __hip_bfloat16* SELB = (__hip_bfloat16*)(ws + alloc((size_t)B_ * DIM_ * 2));
  float*          AOUT = (float*)(ws + alloc((size_t)B_ * DIM_ * 4));
  __hip_bfloat16* AOUTB= (__hip_bfloat16*)(ws + alloc((size_t)B_ * DIM_ * 2));
  float*          PROJ = (float*)(ws + alloc((size_t)B_ * DIM_ * 4));

  // --- weight transposes to bf16 (N,K); wk/wv interleave into [K|V] per layer ---
  transpose_to_bf16<<<dim3((PD_ * DIM_ + 255) / 256, 1, 1), 256, 0, stream>>>(
      pe_w, peT, PD_, DIM_, PD_ * DIM_);
  transpose_to_bf16<<<dim3((DIM_ * DIM_ + 255) / 256, 1, 4), 256, 0, stream>>>(
      wq, wqT, DIM_, DIM_, DIM_ * DIM_);
  transpose_to_bf16<<<dim3((DIM_ * DIM_ + 255) / 256, 1, 4), 256, 0, stream>>>(
      wk, wkvT, DIM_, DIM_, KV_ * DIM_);
  transpose_to_bf16<<<dim3((DIM_ * DIM_ + 255) / 256, 1, 4), 256, 0, stream>>>(
      wv, wkvT + (size_t)DIM_ * DIM_, DIM_, DIM_, KV_ * DIM_);
  transpose_to_bf16<<<dim3((DIM_ * DIM_ + 255) / 256, 1, 4), 256, 0, stream>>>(
      wo, woT, DIM_, DIM_, DIM_ * DIM_);
  transpose_to_bf16<<<dim3((DIM_ * MLP_ + 255) / 256, 1, 4), 256, 0, stream>>>(
      w1, w1T, DIM_, MLP_, DIM_ * MLP_);
  transpose_to_bf16<<<dim3((MLP_ * DIM_ + 255) / 256, 1, 4), 256, 0, stream>>>(
      w2, w2T, MLP_, DIM_, MLP_ * DIM_);

  // --- patch embedding ---
  patch_embed_ln<<<NTOK, 256, 0, stream>>>(bev, av, S, pe_ln1_g, pe_ln1_b, GB);
  gemm_bf16_wmma<1><<<dim3(DIM_ / 128, NTOK / 128), 256, 0, stream>>>(
      GB, peT, pe_b, X, nullptr, NTOK, DIM_, PD_);
  ln2_pos<<<NTOK, 256, 0, stream>>>(X, XB, pe_ln2_g, pe_ln2_b, pos_emb);

  // --- transformer layers ---
  for (int l = 0; l < 4; ++l) {
    const size_t wsq = (size_t)l * DIM_ * DIM_;
    // fused K|V projection (N = 1024)
    gemm_bf16_wmma<2><<<dim3(KV_ / 128, NTOK / 128), 256, 0, stream>>>(
        XB, wkvT + (size_t)l * KV_ * DIM_, nullptr, nullptr, KVB, NTOK, KV_, DIM_);
    gather_sel<<<B_, 256, 0, stream>>>(X, positions, SELB);
    gemm_bf16_wmma<0><<<dim3(DIM_ / 128, B_ / 128), 256, 0, stream>>>(
        SELB, wqT + wsq, nullptr, QF, nullptr, B_, DIM_, DIM_);
    attn_kernel<<<dim3(B_, 8), 256, 0, stream>>>(QF, KVB, AOUT);
    f32_to_bf16_kernel<<<(B_ * DIM_ + 255) / 256, 256, 0, stream>>>(AOUT, AOUTB, B_ * DIM_);
    gemm_bf16_wmma<1><<<dim3(DIM_ / 128, B_ / 128), 256, 0, stream>>>(
        AOUTB, woT + wsq, bo + (size_t)l * DIM_, PROJ, nullptr, B_, DIM_, DIM_);
    scatter_add<<<B_, 256, 0, stream>>>(X, PROJ, positions);
    // FFN (reuse KVB as the bf16 hidden-input buffer; K/V are dead now)
    __hip_bfloat16* HB = KVB;
    ln_to_bf16<<<NTOK, 256, 0, stream>>>(X, HB, ff_ln_g + (size_t)l * DIM_, ff_ln_b + (size_t)l * DIM_);
    gemm_bf16_wmma<3><<<dim3(MLP_ / 128, NTOK / 128), 256, 0, stream>>>(
        HB, w1T + (size_t)l * DIM_ * MLP_, b1 + (size_t)l * MLP_, nullptr, GB, NTOK, MLP_, DIM_);
    gemm_bf16_wmma<4><<<dim3(DIM_ / 128, NTOK / 128), 256, 0, stream>>>(
        GB, w2T + (size_t)l * MLP_ * DIM_, b2 + (size_t)l * DIM_, X, XB, NTOK, DIM_, MLP_);
  }

  // --- final norm + classification head ---
  head_kernel<<<B_, 128, 0, stream>>>(X, norm_g, norm_b, head_w, head_b, out);

  (void)n_in; (void)out_size; (void)ws_size;
}
